// Node2Edge2NodeBlock_5557687681588
// MI455X (gfx1250) — matrix operand order repaired and checked by reading, not compiled, not run on hardware
//
#include <hip/hip_runtime.h>
#include <stdint.h>

// ---------------- problem constants ----------------
#define NODE_DIM 128
#define E_TOT    640000
#define N_NODES  10000
#define K1       384          // 2*node + edge dims
#define KP1      392          // padded LDS row stride (bf16 elems), 16B aligned
#define K2       256          // hidden + node dims
#define KP2      264

#define USE_ASM_CVT 1         // v_cvt_pk_bf16_f32 via inline asm (CDNA5 VALU)

typedef __bf16 bf16x16 __attribute__((ext_vector_type(16)));
typedef float  f32x8   __attribute__((ext_vector_type(8)));

union Frag16 {
    bf16x16        bf;
    unsigned int   u[8];
    unsigned short us[16];
    uint4          q[2];
};

__device__ __forceinline__ unsigned int pkbf(float lo, float hi) {
#if USE_ASM_CVT
    unsigned int r;
    asm("v_cvt_pk_bf16_f32 %0, %1, %2" : "=v"(r) : "v"(lo), "v"(hi));
    return r;
#else
    // round-to-nearest-even f32 -> packed bf16x2 (manual)
    unsigned a = __float_as_uint(lo);
    unsigned b = __float_as_uint(hi);
    a += 0x7FFFu + ((a >> 16) & 1u);
    b += 0x7FFFu + ((b >> 16) & 1u);
    return (a >> 16) | (b & 0xFFFF0000u);
#endif
}

// One K-step (K-slice of 32): A from global f32 (on-the-fly bf16 cvt),
// all 8 B fragments loaded up-front (16 ds_load_b128 in one clause),
// then 8 back-to-back WMMAs (disjoint accumulators, shared A).
template <int KP>
__device__ __forceinline__ void kstep(f32x8 (&acc)[8],
                                      const float* __restrict__ g0,
                                      const unsigned short* __restrict__ lbase) {
    // A fragment (16-bit 16x32 layout): lane half h holds K = 8h+{0..7}, 16+8h+{0..7}
    Frag16 a;
    float4 v0 = *(const float4*)(g0);
    float4 v1 = *(const float4*)(g0 + 4);
    float4 v2 = *(const float4*)(g0 + 16);
    float4 v3 = *(const float4*)(g0 + 20);
    a.u[0] = pkbf(v0.x, v0.y); a.u[1] = pkbf(v0.z, v0.w);
    a.u[2] = pkbf(v1.x, v1.y); a.u[3] = pkbf(v1.z, v1.w);
    a.u[4] = pkbf(v2.x, v2.y); a.u[5] = pkbf(v2.z, v2.w);
    a.u[6] = pkbf(v3.x, v3.y); a.u[7] = pkbf(v3.z, v3.w);

    Frag16 b0[4], b1[4];
#pragma unroll
    for (int j = 0; j < 4; ++j) {
        const unsigned short* wr = lbase + j * (16 * KP);
        b0[j].q[0] = *(const uint4*)(wr);
        b0[j].q[1] = *(const uint4*)(wr + 16);
    }
#pragma unroll
    for (int j = 0; j < 4; ++j) {
        const unsigned short* wr = lbase + (4 + j) * (16 * KP);
        b1[j].q[0] = *(const uint4*)(wr);
        b1[j].q[1] = *(const uint4*)(wr + 16);
    }
#pragma unroll
    for (int j = 0; j < 4; ++j)
        acc[j] = __builtin_amdgcn_wmma_f32_16x16x32_bf16(
            false, a.bf, false, b0[j].bf, (short)0, acc[j], false, false);
#pragma unroll
    for (int j = 0; j < 4; ++j)
        acc[4 + j] = __builtin_amdgcn_wmma_f32_16x16x32_bf16(
            false, a.bf, false, b1[j].bf, (short)0, acc[4 + j], false, false);
}

// Stage W^T into LDS as bf16, packed 2 k-values per dword store.
// W is [K][128] row-major f32; LDS is [n][k] bf16 with row stride KP.
template <int K, int KP>
__device__ __forceinline__ void stageW(unsigned int* ldsw,       // dword view of LDS
                                       const float* __restrict__ W,
                                       int tid) {
    for (int i = tid; i < (K / 2) * NODE_DIM; i += 256) {
        int n  = i & 127;
        int kh = i >> 7;             // 0 .. K/2-1
        int k  = kh * 2;
        float lo = W[(long long)k * NODE_DIM + n];
        float hi = W[(long long)(k + 1) * NODE_DIM + n];
        ldsw[(n * KP) / 2 + kh] = pkbf(lo, hi);
    }
}

// ============ kernel 1: fused gather + edge GEMM + bias + scatter-add ============
__global__ void __launch_bounds__(256) edge_gemm_scatter(
    const float* __restrict__ x,
    const long long* __restrict__ eidx,   // [2, E] int64: row0=src, row1=dst
    const float* __restrict__ ea,
    const float* __restrict__ W1,         // [384, 128] row-major
    const float* __restrict__ b1,
    float* __restrict__ agg)              // [N, 128], pre-zeroed
{
    extern __shared__ __align__(16) unsigned short lds[];   // W1^T bf16: [n][k]
    const int tid = threadIdx.x;
    stageW<K1, KP1>((unsigned int*)lds, W1, tid);
    __syncthreads();

    const int lane = tid & 31;
    const int half = lane >> 4;
    const int ml   = lane & 15;
    const long long* srcI = eidx;
    const long long* dstI = eidx + E_TOT;
    const unsigned short* lB = lds + ml * KP1 + half * 8;

    float bv[8];
#pragma unroll
    for (int nt = 0; nt < 8; ++nt) bv[nt] = b1[nt * 16 + ml];

    const int waveId = blockIdx.x * (blockDim.x >> 5) + (tid >> 5);
    const int nWaves = gridDim.x * (blockDim.x >> 5);
    const int nTiles = E_TOT / 16;

    for (int tile = waveId; tile < nTiles; tile += nWaves) {
        const long long e0 = (long long)tile * 16;
        const long long e  = e0 + ml;
        const float* pS = x + (long long)(int)srcI[e] * NODE_DIM + half * 8;
        const float* pD = x + (long long)(int)dstI[e] * NODE_DIM + half * 8;
        const float* pE = ea + e * NODE_DIM + half * 8;

        // 8 dst indices for this lane-half: 4x b128 loads (int64 low dwords)
        int d8[8];
        {
            const uint4* p = (const uint4*)(dstI + e0 + half * 8);
            uint4 q0 = p[0], q1 = p[1], q2 = p[2], q3 = p[3];
            d8[0] = (int)q0.x; d8[1] = (int)q0.z;
            d8[2] = (int)q1.x; d8[3] = (int)q1.z;
            d8[4] = (int)q2.x; d8[5] = (int)q2.z;
            d8[6] = (int)q3.x; d8[7] = (int)q3.z;
        }

        f32x8 acc[8];
#pragma unroll
        for (int nt = 0; nt < 8; ++nt) acc[nt] = (f32x8)0.0f;

        // segment 0: x[src], K = 0..127
#pragma unroll 1
        for (int t = 0; t < 4; ++t) kstep<KP1>(acc, pS + t * 32, lB + t * 32);
        // segment 1: x[dst], K = 128..255
#pragma unroll 1
        for (int t = 0; t < 4; ++t) kstep<KP1>(acc, pD + t * 32, lB + 128 + t * 32);
        // segment 2: edge_attr, K = 256..383
#pragma unroll 1
        for (int t = 0; t < 4; ++t) kstep<KP1>(acc, pE + t * 32, lB + 256 + t * 32);

        // epilogue: + b1, scatter-add into agg[dst] (L2-resident, hw f32 atomics)
#pragma unroll
        for (int r = 0; r < 8; ++r) {
            float* base = agg + (long long)d8[r] * NODE_DIM + ml;
#pragma unroll
            for (int nt = 0; nt < 8; ++nt)
                unsafeAtomicAdd(base + nt * 16, acc[nt][r] + bv[nt]);
        }
    }
}

// ============ kernel 2: node GEMM  out = [agg | x] @ W2 + b2 ============
__global__ void __launch_bounds__(256) node_gemm(
    const float* __restrict__ agg,
    const float* __restrict__ x,
    const float* __restrict__ W2,         // [256, 128] row-major
    const float* __restrict__ b2,
    float* __restrict__ out)              // [N, 128]
{
    extern __shared__ __align__(16) unsigned short lds[];   // W2^T bf16
    const int tid = threadIdx.x;
    stageW<K2, KP2>((unsigned int*)lds, W2, tid);
    __syncthreads();

    const int lane = tid & 31;
    const int half = lane >> 4;
    const int ml   = lane & 15;
    const unsigned short* lB = lds + ml * KP2 + half * 8;

    float bv[8];
#pragma unroll
    for (int nt = 0; nt < 8; ++nt) bv[nt] = b2[nt * 16 + ml];

    const int waveId = blockIdx.x * (blockDim.x >> 5) + (tid >> 5);
    const int nWaves = gridDim.x * (blockDim.x >> 5);
    const int nTiles = N_NODES / 16;

    for (int tile = waveId; tile < nTiles; tile += nWaves) {
        const int node = tile * 16 + ml;
        const float* p0 = agg + (long long)node * NODE_DIM + half * 8;
        const float* p1 = x   + (long long)node * NODE_DIM + half * 8;

        f32x8 acc[8];
#pragma unroll
        for (int nt = 0; nt < 8; ++nt) acc[nt] = (f32x8)0.0f;

        // segment 0: agg, K = 0..127
#pragma unroll 1
        for (int t = 0; t < 4; ++t) kstep<KP2>(acc, p0 + t * 32, lB + t * 32);
        // segment 1: x, K = 128..255
#pragma unroll 1
        for (int t = 0; t < 4; ++t) kstep<KP2>(acc, p1 + t * 32, lB + 128 + t * 32);

#pragma unroll
        for (int r = 0; r < 8; ++r) {
            float* base = out + ((long long)tile * 16 + r + 8 * half) * NODE_DIM + ml;
#pragma unroll
            for (int nt = 0; nt < 8; ++nt)
                base[nt * 16] = acc[nt][r] + bv[nt];
        }
    }
}

// ---------------- host launcher ----------------
extern "C" void kernel_launch(void* const* d_in, const int* in_sizes, int n_in,
                              void* d_out, int out_size, void* d_ws, size_t ws_size,
                              hipStream_t stream) {
    const float*     x    = (const float*)d_in[0];
    const long long* eidx = (const long long*)d_in[1];   // int64 per reference
    const float*     ea   = (const float*)d_in[2];
    const float*     W1   = (const float*)d_in[3];
    const float*     b1   = (const float*)d_in[4];
    const float*     W2   = (const float*)d_in[5];
    const float*     b2   = (const float*)d_in[6];
    float* agg = (float*)d_ws;                           // [N, 128] scratch
    float* out = (float*)d_out;

    const size_t smem1 = (size_t)NODE_DIM * KP1 * sizeof(unsigned short); // ~98 KB
    const size_t smem2 = (size_t)NODE_DIM * KP2 * sizeof(unsigned short); // ~66 KB
    (void)hipFuncSetAttribute((const void*)edge_gemm_scatter,
                              hipFuncAttributeMaxDynamicSharedMemorySize, (int)smem1);
    (void)hipFuncSetAttribute((const void*)node_gemm,
                              hipFuncAttributeMaxDynamicSharedMemorySize, (int)smem2);

    (void)hipMemsetAsync(agg, 0, (size_t)N_NODES * NODE_DIM * sizeof(float), stream);
    edge_gemm_scatter<<<512, 256, smem1, stream>>>(x, eidx, ea, W1, b1, agg);
    node_gemm<<<80, 256, smem2, stream>>>(agg, x, W2, b2, out);
}
